// MultiHeadAttention_68771016343804
// MI455X (gfx1250) — matrix operand order repaired and checked by reading, not compiled, not run on hardware
//
#include <hip/hip_runtime.h>
#include <hip/hip_bf16.h>

#define DIM       1024
#define NUM_HEADS 16
#define HEAD_DIM  64
#define BATCH     2
#define SEQ       2048

typedef __attribute__((ext_vector_type(16))) _Float16 v16h;
typedef __attribute__((ext_vector_type(8)))  _Float16 v8h;
typedef __attribute__((ext_vector_type(8)))  float    v8f;
typedef __attribute__((ext_vector_type(4)))  int      v4i;

// ---- CDNA5 async global->LDS support (guarded; falls back to ds stores) ----
#if defined(__has_builtin)
#  if __has_builtin(__builtin_amdgcn_global_load_async_to_lds_b128)
#    define HAVE_ASYNC_LDS 1
#  else
#    define HAVE_ASYNC_LDS 0
#  endif
#  if __has_builtin(__builtin_amdgcn_s_wait_asynccnt)
#    define HAVE_WAIT_ASYNC 1
#  else
#    define HAVE_WAIT_ASYNC 0
#  endif
#else
#  define HAVE_ASYNC_LDS 0
#  define HAVE_WAIT_ASYNC 0
#endif

#if HAVE_ASYNC_LDS
typedef v4i __attribute__((address_space(1)))* gbl_v4i_p;  // global
typedef v4i __attribute__((address_space(3)))* lds_v4i_p;  // LDS

static __device__ __forceinline__ void async_copy_b128(const _Float16* g, _Float16* l) {
  __builtin_amdgcn_global_load_async_to_lds_b128(
      (gbl_v4i_p)g, (lds_v4i_p)l, /*imm offset=*/0, /*cpol=*/0);
}
#endif

static __device__ __forceinline__ void wait_async0() {
#if HAVE_ASYNC_LDS
#  if HAVE_WAIT_ASYNC
  __builtin_amdgcn_s_wait_asynccnt(0);
#  else
  asm volatile("s_wait_asynccnt 0x0" ::: "memory");
#  endif
#endif
}

static __device__ __forceinline__ v8f vzero8() {
  v8f z;
#pragma unroll
  for (int i = 0; i < 8; ++i) z[i] = 0.0f;
  return z;
}

static __device__ __forceinline__ v8f wmma_f16(v16h a, v16h b, v8f c) {
  // D = A(16x32 f16) * B(32x16 f16) + C(16x16 f32)
  return __builtin_amdgcn_wmma_f32_16x16x32_f16(
      /*neg_a=*/false, a, /*neg_b=*/false, b,
      /*c_mod=*/(short)0, c, /*reuse_a=*/false, /*reuse_b=*/false);
}

// A operand 16x32 f16, contiguous k order: lanes 0-15 (row M = lane) hold K={0..15},
// lanes 16-31 (row M = lane-16) hold K={16..31}. half8 = (lane>>4)*8.
static __device__ __forceinline__ v16h load_a(const _Float16* rowPtr, int kb, int half8) {
  v8h r0 = *(const v8h*)(rowPtr + kb + half8 * 2);        // VGPRs 0..3
  v8h r1 = *(const v8h*)(rowPtr + kb + half8 * 2 + 8);   // VGPRs 4..7
  return __builtin_shufflevector(r0, r1, 0,1,2,3,4,5,6,7,8,9,10,11,12,13,14,15);
}

// ---------------------------------------------------------------------------
// Kernel 0: fp32 -> f16 conversion of x and the four weight matrices
// ---------------------------------------------------------------------------
__global__ void __launch_bounds__(256)
convert_f16_kernel(const float* __restrict__ x,  const float* __restrict__ wq,
                   const float* __restrict__ wk, const float* __restrict__ wv,
                   const float* __restrict__ wo,
                   _Float16* xf, _Float16* wqf, _Float16* wkf,
                   _Float16* wvf, _Float16* wof) {
  const size_t NX = (size_t)BATCH * SEQ * DIM;   // 4,194,304
  const size_t NW = (size_t)DIM * DIM;           // 1,048,576
  size_t i = (size_t)blockIdx.x * 256 + threadIdx.x;
  if (i < NX) {
    xf[i] = (_Float16)x[i];
  } else {
    size_t r = i - NX;
    size_t w = r / NW, o = r - w * NW;
    if (w == 0)      wqf[o] = (_Float16)wq[o];
    else if (w == 1) wkf[o] = (_Float16)wk[o];
    else if (w == 2) wvf[o] = (_Float16)wv[o];
    else if (w == 3) wof[o] = (_Float16)wo[o];
  }
}

// ---------------------------------------------------------------------------
// Shared-B GEMM body: 4 waves x (16 rows x 64 cols). B tile (64x32 f16 = 4KB)
// staged per block in double-buffered LDS via async global->LDS copies.
// ---------------------------------------------------------------------------
struct BStage {
  _Float16 (*buf)[64 * 32];          // [2][2048]
  const _Float16* wRow;              // this thread's source row segment
  int crow, cseg;

  __device__ __forceinline__ void init(_Float16 (*b)[64 * 32],
                                       const _Float16* w, int colBase, int tid) {
    buf  = b;
    crow = tid >> 1;                 // 0..63 : output column within 64-wide band
    cseg = (tid & 1) * 16;           // first/second 16 halves of the 32-wide k chunk
    wRow = w + (size_t)(colBase + crow) * DIM + cseg;
  }
  __device__ __forceinline__ void issue(int p, int kb) {
#if HAVE_ASYNC_LDS
    async_copy_b128(wRow + kb,     &buf[p][crow * 32 + cseg]);
    async_copy_b128(wRow + kb + 8, &buf[p][crow * 32 + cseg + 8]);
#else
    v8h t0 = *(const v8h*)(wRow + kb);
    v8h t1 = *(const v8h*)(wRow + kb + 8);
    *(v8h*)&buf[p][crow * 32 + cseg]     = t0;
    *(v8h*)&buf[p][crow * 32 + cseg + 8] = t1;
#endif
  }
};

// ---------------------------------------------------------------------------
// Kernel 1: QKV projections + RoPE epilogue (Q,K) / transpose epilogue (V)
//   grid = (64, NUM_HEADS, 3), block = 128 (4 waves).
// ---------------------------------------------------------------------------
__global__ void __launch_bounds__(128)
qkv_proj_kernel(const _Float16* __restrict__ xf,
                const _Float16* __restrict__ wqf,
                const _Float16* __restrict__ wkf,
                const _Float16* __restrict__ wvf,
                _Float16* __restrict__ qf,    // [B,S,H,D]
                _Float16* __restrict__ kf,    // [B,S,H,D]
                _Float16* __restrict__ vtf) { // [B,H,D,S]
  const int lane  = threadIdx.x & 31;
  const int wave  = threadIdx.x >> 5;
  const int l16   = lane & 15;
  const int half8 = (lane >> 4) * 8;

  const int rowBase = (blockIdx.x * 4 + wave) * 16;  // token index (b*S+s)
  const int head    = blockIdx.y;
  const int colBase = head * HEAD_DIM;
  const int mode    = blockIdx.z;                    // 0=Q 1=K 2=V

  const _Float16* w = (mode == 0) ? wqf : (mode == 1) ? wkf : wvf;
  const _Float16* aRow = xf + (size_t)(rowBase + l16) * DIM;

  __shared__ __align__(16) _Float16 bbuf[2][64 * 32];
  BStage st;
  st.init(bbuf, w, colBase, threadIdx.x);

  v8f acc[4];
#pragma unroll
  for (int t = 0; t < 4; ++t) acc[t] = vzero8();

  st.issue(0, 0);
  wait_async0();
  __syncthreads();

  for (int kb = 0; kb < DIM; kb += 32) {
    const int p = (kb >> 5) & 1;
    if (kb + 32 < DIM) st.issue(p ^ 1, kb + 32);   // prefetch next B tile
    v16h a = load_a(aRow, kb, half8);
    __builtin_prefetch(aRow + kb + 64, 0, 0);
#pragma unroll
    for (int t = 0; t < 4; ++t) {
      v16h b = *(const v16h*)&bbuf[p][(16 * t + l16) * 32 + half8 * 2];
      acc[t] = wmma_f16(a, b, acc[t]);
    }
    wait_async0();
    __syncthreads();
  }

  if (mode == 2) {
    // V: write transposed f16 [B,H,D,S]
#pragma unroll
    for (int t = 0; t < 4; ++t) {
      const int d = 16 * t + l16;
#pragma unroll
      for (int j = 0; j < 8; ++j) {
        const int row = rowBase + j + half8;      // b*S + s
        const int b = row >> 11;                  // /SEQ
        const int s = row & (SEQ - 1);
        vtf[(((size_t)b * NUM_HEADS + head) * HEAD_DIM + d) * SEQ + s] =
            (_Float16)acc[t][j];
      }
    }
  } else {
    _Float16* dst = (mode == 0) ? qf : kf;
#pragma unroll
    for (int t = 0; t < 4; ++t) {
      const int d = 16 * t + l16;
      const float invf = __powf(10000.0f, -(float)(d & 31) * (1.0f / 32.0f));
#pragma unroll
      for (int j = 0; j < 8; ++j) {
        const int row = rowBase + j + half8;
        const int s = row & (SEQ - 1);
        float sn, cs;
        __sincosf((float)s * invf, &sn, &cs);
        const float other = acc[t ^ 2][j];        // q[d +/- 32] lives in tile t^2
        const float rot = (d < 32) ? -other : other;
        dst[(size_t)row * DIM + colBase + d] = (_Float16)(acc[t][j] * cs + rot * sn);
      }
    }
  }
}

// ---------------------------------------------------------------------------
// Kernel 2: flash attention. grid = (SEQ/16, NUM_HEADS, BATCH), block = 32.
// ---------------------------------------------------------------------------
__global__ void __launch_bounds__(32)
attn_kernel(const _Float16* __restrict__ qf,   // [B,S,H,D]
            const _Float16* __restrict__ kf,   // [B,S,H,D]
            const _Float16* __restrict__ vtf,  // [B,H,D,S]
            _Float16* __restrict__ aof) {      // [B,S,H*D]
  const int lane  = threadIdx.x & 31;
  const int l16   = lane & 15;
  const int half8 = (lane >> 4) * 8;

  const int qt   = blockIdx.x;
  const int head = blockIdx.y;
  const int b    = blockIdx.z;

  __shared__ __align__(16) _Float16 pbuf[16 * 32];

  const _Float16* qrow =
      qf + (size_t)(b * SEQ + qt * 16 + l16) * DIM + head * HEAD_DIM;
  v16h qa[2];
  qa[0] = load_a(qrow, 0, half8);
  qa[1] = load_a(qrow, 32, half8);

  v8f o[4];
#pragma unroll
  for (int t = 0; t < 4; ++t) o[t] = vzero8();

  float mrow[8], lrow[8];
#pragma unroll
  for (int j = 0; j < 8; ++j) { mrow[j] = -1e30f; lrow[j] = 0.0f; }

  const float scale = 0.125f;  // HEAD_DIM^-0.5
  const _Float16* vbase = vtf + (size_t)(b * NUM_HEADS + head) * HEAD_DIM * SEQ;

  for (int kb = 0; kb < SEQ; kb += 32) {
    v8f s0 = vzero8(), s1 = vzero8();
    const _Float16* kbase =
        kf + (size_t)(b * SEQ + kb) * DIM + head * HEAD_DIM;
#pragma unroll
    for (int ks = 0; ks < 2; ++ks) {
      v16h b0 = *(const v16h*)(kbase + (size_t)l16 * DIM + 32 * ks + half8 * 2);
      v16h b1 = *(const v16h*)(kbase + (size_t)(16 + l16) * DIM + 32 * ks + half8 * 2);
      s0 = wmma_f16(qa[ks], b0, s0);
      s1 = wmma_f16(qa[ks], b1, s1);
    }

    float corr[8];
#pragma unroll
    for (int j = 0; j < 8; ++j) {
      const float v0 = s0[j] * scale;
      const float v1 = s1[j] * scale;
      float rm = fmaxf(v0, v1);                 // reduce over n within 16-lane group
      rm = fmaxf(rm, __shfl_xor(rm, 1, 32));
      rm = fmaxf(rm, __shfl_xor(rm, 2, 32));
      rm = fmaxf(rm, __shfl_xor(rm, 4, 32));
      rm = fmaxf(rm, __shfl_xor(rm, 8, 32));
      const float mn = fmaxf(mrow[j], rm);
      const float c  = __expf(mrow[j] - mn);
      mrow[j] = mn;
      const float p0 = __expf(v0 - mn);
      const float p1 = __expf(v1 - mn);
      float rs = p0 + p1;
      rs += __shfl_xor(rs, 1, 32);
      rs += __shfl_xor(rs, 2, 32);
      rs += __shfl_xor(rs, 4, 32);
      rs += __shfl_xor(rs, 8, 32);
      lrow[j] = lrow[j] * c + rs;
      corr[j] = c;
      const int m = j + half8;                  // D-layout row
      pbuf[m * 32 + l16]      = (_Float16)p0;
      pbuf[m * 32 + 16 + l16] = (_Float16)p1;
    }
#pragma unroll
    for (int ds = 0; ds < 4; ++ds)
#pragma unroll
      for (int j = 0; j < 8; ++j) o[ds][j] *= corr[j];

    __syncthreads();
    v8h r0 = *(const v8h*)(pbuf + l16 * 32 + half8 * 2);
    v8h r1 = *(const v8h*)(pbuf + l16 * 32 + half8 * 2 + 8);
    v16h pa = __builtin_shufflevector(r0, r1,
        0,1,2,3,4,5,6,7,8,9,10,11,12,13,14,15);
#pragma unroll
    for (int ds = 0; ds < 4; ++ds) {
      v16h vb = *(const v16h*)(vbase + (size_t)(ds * 16 + l16) * SEQ + kb + half8 * 2);
      o[ds] = wmma_f16(pa, vb, o[ds]);
    }
    __syncthreads();
  }

#pragma unroll
  for (int ds = 0; ds < 4; ++ds)
#pragma unroll
    for (int j = 0; j < 8; ++j) {
      const int sq = qt * 16 + j + half8;
      aof[(size_t)(b * SEQ + sq) * DIM + head * HEAD_DIM + ds * 16 + l16] =
          (_Float16)(o[ds][j] / lrow[j]);
    }
}

// ---------------------------------------------------------------------------
// Kernel 3: output projection, f32 result. grid = (64, DIM/64), block = 128.
// ---------------------------------------------------------------------------
__global__ void __launch_bounds__(128)
out_proj_kernel(const _Float16* __restrict__ af,
                const _Float16* __restrict__ wof,
                float* __restrict__ out) {
  const int lane  = threadIdx.x & 31;
  const int wave  = threadIdx.x >> 5;
  const int l16   = lane & 15;
  const int half8 = (lane >> 4) * 8;

  const int rowBase = (blockIdx.x * 4 + wave) * 16;
  const int colBase = blockIdx.y * 64;

  const _Float16* aRow = af + (size_t)(rowBase + l16) * DIM;

  __shared__ __align__(16) _Float16 bbuf[2][64 * 32];
  BStage st;
  st.init(bbuf, wof, colBase, threadIdx.x);

  v8f acc[4];
#pragma unroll
  for (int t = 0; t < 4; ++t) acc[t] = vzero8();

  st.issue(0, 0);
  wait_async0();
  __syncthreads();

  for (int kb = 0; kb < DIM; kb += 32) {
    const int p = (kb >> 5) & 1;
    if (kb + 32 < DIM) st.issue(p ^ 1, kb + 32);
    v16h a = load_a(aRow, kb, half8);
#pragma unroll
    for (int t = 0; t < 4; ++t) {
      v16h b = *(const v16h*)&bbuf[p][(16 * t + l16) * 32 + half8 * 2];
      acc[t] = wmma_f16(a, b, acc[t]);
    }
    wait_async0();
    __syncthreads();
  }

#pragma unroll
  for (int t = 0; t < 4; ++t)
#pragma unroll
    for (int j = 0; j < 8; ++j) {
      const int row = rowBase + j + half8;
      out[(size_t)row * DIM + colBase + 16 * t + l16] = acc[t][j];
    }
}

// ---------------------------------------------------------------------------
extern "C" void kernel_launch(void* const* d_in, const int* in_sizes, int n_in,
                              void* d_out, int out_size, void* d_ws, size_t ws_size,
                              hipStream_t stream) {
  const float* x  = (const float*)d_in[0];
  const float* wq = (const float*)d_in[1];
  const float* wk = (const float*)d_in[2];
  const float* wv = (const float*)d_in[3];
  const float* wo = (const float*)d_in[4];

  const size_t NX = (size_t)BATCH * SEQ * DIM;  // 4,194,304
  const size_t NW = (size_t)DIM * DIM;          // 1,048,576

  _Float16* ws    = (_Float16*)d_ws;
  _Float16* xf    = ws;
  _Float16* wqf   = xf  + NX;
  _Float16* wkf   = wqf + NW;
  _Float16* wvf   = wkf + NW;
  _Float16* wof   = wvf + NW;
  _Float16* qf    = wof + NW;
  _Float16* kfp   = qf  + NX;
  _Float16* vtf   = kfp + NX;
  _Float16* attnf = vtf + NX;

  {
    const size_t total = NX + 4 * NW;           // 8,388,608
    convert_f16_kernel<<<(unsigned)((total + 255) / 256), 256, 0, stream>>>(
        x, wq, wk, wv, wo, xf, wqf, wkf, wvf, wof);
  }
  qkv_proj_kernel<<<dim3(64, NUM_HEADS, 3), 128, 0, stream>>>(
      xf, wqf, wkf, wvf, qf, kfp, vtf);
  attn_kernel<<<dim3(SEQ / 16, NUM_HEADS, BATCH), 32, 0, stream>>>(
      qf, kfp, vtf, attnf);
  out_proj_kernel<<<dim3(64, DIM / 64), 128, 0, stream>>>(
      attnf, wof, (float*)d_out);
}
